// Model_11063835755189
// MI455X (gfx1250) — matrix operand order repaired and checked by reading, not compiled, not run on hardware
//
#include <hip/hip_runtime.h>
#include <hip/hip_bf16.h>
#include <math.h>

typedef _Float16 half_t;
typedef __attribute__((ext_vector_type(16))) _Float16 v16h;
typedef __attribute__((ext_vector_type(8)))  _Float16 v8h;
typedef __attribute__((ext_vector_type(8)))  float    v8f;

#define N_DET   128
#define N_IMG   16
#define MAX_PPL 8
#define RES     32
#define HW      1024
#define CTX     1280
#define DIM     1024
#define HEADS   8
#define DHEAD   64
#define INNER   512
#define MLP     1024
#define NROT    53
#define NPOSE   318
#define TOKDIM  1611
#define TOKPAD  1632
#define OUTD    500

// ---------------------------------------------------------------------------
// prep: rank-within-image, counts, ends (idx_0 is sorted)
// ---------------------------------------------------------------------------
__global__ void prep_kernel(const int* __restrict__ idx0, int* __restrict__ posr,
                            int* __restrict__ counts, int* __restrict__ ends) {
  __shared__ int cnt[N_IMG];
  int n = threadIdx.x;
  if (n < N_IMG) cnt[n] = 0;
  __syncthreads();
  int img = idx0[n];
  int p = 0;
  for (int m = 0; m < n; ++m)
    if (idx0[m] == img) ++p;
  posr[n] = p;
  atomicAdd(&cnt[img], 1);
  __syncthreads();
  if (n == 0) {
    int acc = 0;
    for (int i = 0; i < N_IMG; ++i) { acc += cnt[i]; ends[i] = acc; counts[i] = cnt[i]; }
  }
}

// ---------------------------------------------------------------------------
// f32 -> f16 weight conversion (zero-fills [n_src, n_total) for K padding)
// ---------------------------------------------------------------------------
__global__ void convert_f16(half_t* __restrict__ dst, const float* __restrict__ src,
                            long n_src, long n_total) {
  long i = (long)blockIdx.x * blockDim.x + threadIdx.x;
  if (i < n_total) dst[i] = (i < n_src) ? (half_t)src[i] : (half_t)0.0f;
}

__global__ void combine_bias(float* __restrict__ dst, const float* __restrict__ b,
                             const float* __restrict__ pos) {
  int i = blockIdx.x * blockDim.x + threadIdx.x;
  if (i < DIM) dst[i] = b[i] + pos[i];
}

// ---------------------------------------------------------------------------
// token builder: [x_central + cq_x[iy] + cq_y[ix] | init_pose | init_betas |
//                 init_cam | zero-pad] -> f16 row at (img*8 + rank)
// ---------------------------------------------------------------------------
__global__ void build_token(const float* __restrict__ xc, const float* __restrict__ cqx,
                            const float* __restrict__ cqy, const int* __restrict__ idx0,
                            const int* __restrict__ idxy, const int* __restrict__ idxx,
                            const int* __restrict__ posr,
                            const float* __restrict__ initp, const float* __restrict__ initb,
                            const float* __restrict__ initc, half_t* __restrict__ token) {
  int d = blockIdx.x;
  int row = idx0[d] * MAX_PPL + posr[d];
  int iy = idxy[d], ix = idxx[d];
  half_t* trow = token + (size_t)row * TOKPAD;
  for (int c = threadIdx.x; c < TOKPAD; c += blockDim.x) {
    float v;
    if (c < CTX)                      v = xc[(size_t)d * CTX + c] + cqx[iy * CTX + c] + cqy[ix * CTX + c];
    else if (c < CTX + NPOSE)         v = initp[c - CTX];
    else if (c < CTX + NPOSE + 10)    v = initb[c - CTX - NPOSE];
    else if (c < TOKDIM)              v = initc[c - CTX - NPOSE - 10];
    else                              v = 0.0f;
    trow[c] = (half_t)v;
  }
}

// ---------------------------------------------------------------------------
// ctx builder: gather per-image feature map x[ends-1], add cv at detection
// spots, transpose to [img][hw][c], store f16
// ---------------------------------------------------------------------------
__global__ void build_ctx(const float* __restrict__ x, const float* __restrict__ cvx,
                          const float* __restrict__ cvy, const int* __restrict__ idxy,
                          const int* __restrict__ idxx, const int* __restrict__ counts,
                          const int* __restrict__ ends, half_t* __restrict__ ctx) {
  int img = blockIdx.x;      // 16
  int c   = blockIdx.y;      // 1280
  int hw  = threadIdx.x;     // 1024
  int ed = ends[img] - 1;
  float v = x[((size_t)ed * CTX + c) * HW + hw];
  int dend = ends[img];
  int dstart = dend - counts[img];
  for (int d = dstart; d < dend; ++d) {
    int iy = idxy[d], ix = idxx[d];
    if (iy * RES + ix == hw)
      v += cvx[iy * CTX + c] + cvy[ix * CTX + c];
  }
  ctx[((size_t)img * HW + hw) * CTX + c] = (half_t)v;
}

// ---------------------------------------------------------------------------
// layernorm over DIM, f32 in -> f16 out (WMMA A operand)
// ---------------------------------------------------------------------------
__global__ void layernorm_f16(const float* __restrict__ t, const float* __restrict__ g,
                              const float* __restrict__ b, half_t* __restrict__ out) {
  int row = blockIdx.x;
  const float* tr = t + (size_t)row * DIM;
  __shared__ float redA[8], redB[8];
  float s = 0.f, s2 = 0.f;
  for (int c = threadIdx.x; c < DIM; c += blockDim.x) {
    float v = tr[c]; s += v; s2 += v * v;
  }
  for (int off = 16; off > 0; off >>= 1) {
    s  += __shfl_down(s,  off, 32);
    s2 += __shfl_down(s2, off, 32);
  }
  int wid = threadIdx.x >> 5, lane = threadIdx.x & 31;
  if (lane == 0) { redA[wid] = s; redB[wid] = s2; }
  __syncthreads();
  if (threadIdx.x == 0) {
    float a = 0.f, c2 = 0.f;
    int nw = blockDim.x >> 5;
    for (int i = 0; i < nw; ++i) { a += redA[i]; c2 += redB[i]; }
    redA[0] = a; redB[0] = c2;
  }
  __syncthreads();
  float mean = redA[0] * (1.0f / DIM);
  float var  = redB[0] * (1.0f / DIM) - mean * mean;
  float rstd = rsqrtf(var + 1e-5f);
  for (int c = threadIdx.x; c < DIM; c += blockDim.x) {
    float v = (tr[c] - mean) * rstd * g[c] + b[c];
    out[(size_t)row * DIM + c] = (half_t)v;
  }
}

// ---------------------------------------------------------------------------
// generic WMMA GEMM: C[M,N] = A[M,K]f16 * B[K,N]f16  (+bias, gelu, residual)
// Block = 4 waves covering a 32(M) x 128(N) macro-tile; each wave owns a
// 32x32 register tile = 2x2 WMMA tiles: per K-step, 2 A-frag + 2 B-frag
// loads feed 4 independent v_wmma_f32_16x16x32_f16 (good load/WMMA overlap).
// Addressing is strength-reduced to pointer increments (no in-loop mul64).
// Fragment mapping per CDNA5 ISA 7.12.2:
//   A: lane row = lane&15, K halves {kb..kb+7, 16+kb..16+kb+7}, kb=(lane>>4)*8
//   B: lane col = lane&15, K order same as A (strided gather)
//   C: elem r <-> M = r + 8*(lane>>4), N = lane&15
// Requires: M%32==0, N%32==0, K%32==0 (buffers padded accordingly).
// ---------------------------------------------------------------------------
__global__ void wmma_gemm(const half_t* __restrict__ A, const half_t* __restrict__ B,
                          const float* __restrict__ bias, const float* __restrict__ residual,
                          float* __restrict__ Cf, half_t* __restrict__ Ch,
                          int M, int N, int K, int do_gelu) {
  int wave = threadIdx.x >> 5;
  int lane = threadIdx.x & 31;
  int m0 = blockIdx.x * 32;
  int n0 = blockIdx.y * 128 + wave * 32;
  if (n0 >= N) return;

  int row   = lane & 15;
  int khalf = (lane >> 4) * 8;

  const half_t* ap0 = A + (size_t)(m0 + row) * K + khalf;
  const half_t* ap1 = ap0 + (size_t)16 * K;
  const half_t* bp  = B + (size_t)khalf * N + n0 + row;
  const size_t bstep = (size_t)32 * N;

  v8f acc00 = {}, acc01 = {}, acc10 = {}, acc11 = {};
  for (int k = 0; k < K; k += 32) {
    __builtin_prefetch(ap0 + 64, 0, 1);
    __builtin_prefetch(bp + 2 * bstep, 0, 1);
    v8h a0lo = *(const v8h*)(ap0);
    v8h a0hi = *(const v8h*)(ap0 + 16);
    v8h a1lo = *(const v8h*)(ap1);
    v8h a1hi = *(const v8h*)(ap1 + 16);
    v16h a0 = __builtin_shufflevector(a0lo, a0hi, 0, 1, 2, 3, 4, 5, 6, 7,
                                      8, 9, 10, 11, 12, 13, 14, 15);
    v16h a1 = __builtin_shufflevector(a1lo, a1hi, 0, 1, 2, 3, 4, 5, 6, 7,
                                      8, 9, 10, 11, 12, 13, 14, 15);
    v16h b0, b1;
    for (int s = 0; s < 16; ++s) {
      size_t bo = (size_t)(s + (s & 8)) * N;
      b0[s] = bp[bo];
      b1[s] = bp[bo + 16];
    }
    acc00 = __builtin_amdgcn_wmma_f32_16x16x32_f16(false, a0, false, b0, (short)0, acc00, false, false);
    acc01 = __builtin_amdgcn_wmma_f32_16x16x32_f16(false, a0, false, b1, (short)0, acc01, false, false);
    acc10 = __builtin_amdgcn_wmma_f32_16x16x32_f16(false, a1, false, b0, (short)0, acc10, false, false);
    acc11 = __builtin_amdgcn_wmma_f32_16x16x32_f16(false, a1, false, b1, (short)0, acc11, false, false);
    ap0 += 32; ap1 += 32; bp += bstep;
  }

  int cl = lane & 15;
  int ch = 8 * (lane >> 4);
  auto epi = [&](const v8f& acc, int mbase, int nbase) {
    int gn = nbase + cl;
    float bv = bias ? bias[gn] : 0.0f;
    for (int r = 0; r < 8; ++r) {
      int gm = mbase + ch + r;
      float v = acc[r] + bv;
      if (do_gelu) {
        float x3 = v * v * v;
        v = 0.5f * v * (1.0f + tanhf(0.7978845608028654f * (v + 0.044715f * x3)));
      }
      if (residual) v += residual[(size_t)gm * N + gn];
      if (Cf) Cf[(size_t)gm * N + gn] = v;
      if (Ch) Ch[(size_t)gm * N + gn] = (half_t)v;
    }
  };
  epi(acc00, m0,      n0);
  epi(acc01, m0,      n0 + 16);
  epi(acc10, m0 + 16, n0);
  epi(acc11, m0 + 16, n0 + 16);
}

// ---------------------------------------------------------------------------
// self-attention: per (img, head), 8 tokens, key mask from counts
// qkv layout: [img*8+p][1536] = [q(512) | k(512) | v(512)], head slices of 64
// ---------------------------------------------------------------------------
__global__ void self_attn(const half_t* __restrict__ qkv, const int* __restrict__ counts,
                          half_t* __restrict__ out) {
  int img = blockIdx.x >> 3;
  int h   = blockIdx.x & 7;
  int t   = threadIdx.x;  // 64 threads
  __shared__ float sc[MAX_PPL][MAX_PPL];
  __shared__ float pr[MAX_PPL][MAX_PPL];
  int cnt = counts[img];
  int qi = t >> 3, ki = t & 7;
  const half_t* base = qkv + (size_t)img * MAX_PPL * (3 * INNER);
  float s = 0.0f;
  for (int d = 0; d < DHEAD; ++d)
    s += (float)base[qi * 1536 + h * DHEAD + d] *
         (float)base[ki * 1536 + INNER + h * DHEAD + d];
  s *= 0.125f;  // DHEAD^-0.5
  if (ki >= cnt) s = -1e9f;
  sc[qi][ki] = s;
  __syncthreads();
  float mx = -1e30f;
  for (int k = 0; k < MAX_PPL; ++k) mx = fmaxf(mx, sc[qi][k]);
  float den = 0.0f;
  for (int k = 0; k < MAX_PPL; ++k) den += __expf(sc[qi][k] - mx);
  pr[qi][ki] = __expf(s - mx) / den;
  __syncthreads();
  for (int idx = t; idx < MAX_PPL * DHEAD; idx += 64) {
    int q2 = idx >> 6, d = idx & 63;
    float o = 0.0f;
    for (int k = 0; k < MAX_PPL; ++k)
      o += pr[q2][k] * (float)base[k * 1536 + 2 * INNER + h * DHEAD + d];
    out[((size_t)img * MAX_PPL + q2) * INNER + h * DHEAD + d] = (half_t)o;
  }
}

// ---------------------------------------------------------------------------
// cross-attention: per (img, head), 8 queries x 1024 keys; kv layout
// [img*1024+n][1024] = [k(512) | v(512)]
// ---------------------------------------------------------------------------
__global__ void cross_attn(const half_t* __restrict__ q, const half_t* __restrict__ kv,
                           half_t* __restrict__ out) {
  int img = blockIdx.x >> 3;
  int h   = blockIdx.x & 7;
  __shared__ float prob[MAX_PPL * HW];  // 32 KB LDS
  const half_t* qb  = q  + (size_t)img * MAX_PPL * INNER + h * DHEAD;
  const half_t* kvb = kv + (size_t)img * HW * (2 * INNER);
  for (int idx = threadIdx.x; idx < MAX_PPL * HW; idx += blockDim.x) {
    int qi = idx >> 10, n = idx & (HW - 1);
    const half_t* krow = kvb + (size_t)n * (2 * INNER) + h * DHEAD;
    float s = 0.0f;
    for (int d = 0; d < DHEAD; ++d)
      s += (float)qb[(size_t)qi * INNER + d] * (float)krow[d];
    prob[idx] = s * 0.125f;
  }
  __syncthreads();
  int wid = threadIdx.x >> 5, lane = threadIdx.x & 31;  // 8 waves <-> 8 query rows
  {
    float mx = -1e30f;
    for (int n = lane; n < HW; n += 32) mx = fmaxf(mx, prob[wid * HW + n]);
    for (int off = 16; off > 0; off >>= 1) mx = fmaxf(mx, __shfl_xor(mx, off, 32));
    float den = 0.0f;
    for (int n = lane; n < HW; n += 32) {
      float e = __expf(prob[wid * HW + n] - mx);
      prob[wid * HW + n] = e;
      den += e;
    }
    for (int off = 16; off > 0; off >>= 1) den += __shfl_xor(den, off, 32);
    float inv = 1.0f / den;
    for (int n = lane; n < HW; n += 32) prob[wid * HW + n] *= inv;
  }
  __syncthreads();
  for (int idx = threadIdx.x; idx < MAX_PPL * DHEAD; idx += blockDim.x) {
    int qi = idx >> 6, d = idx & 63;
    const half_t* vb = kvb + INNER + h * DHEAD + d;
    float o = 0.0f;
    for (int n = 0; n < HW; ++n)
      o += prob[qi * HW + n] * (float)vb[(size_t)n * (2 * INNER)];
    out[((size_t)img * MAX_PPL + qi) * INNER + h * DHEAD + d] = (half_t)o;
  }
}

// ---------------------------------------------------------------------------
// decode: per detection, 4 linear heads + rot6d -> [128,500]
// ---------------------------------------------------------------------------
__global__ void decode_kernel(const float* __restrict__ t, const int* __restrict__ idx0,
                              const int* __restrict__ posr,
                              const float* __restrict__ wp, const float* __restrict__ bp,
                              const float* __restrict__ ip,
                              const float* __restrict__ wsh, const float* __restrict__ bsh,
                              const float* __restrict__ ib,
                              const float* __restrict__ wc, const float* __restrict__ bc,
                              const float* __restrict__ ic,
                              const float* __restrict__ we, const float* __restrict__ be,
                              const float* __restrict__ ie,
                              float* __restrict__ out) {
  int d = blockIdx.x;
  __shared__ float tok[DIM];
  __shared__ float pose[NPOSE];
  __shared__ float betas[10], cam[3], expr[10];
  int row = idx0[d] * MAX_PPL + posr[d];
  for (int c = threadIdx.x; c < DIM; c += blockDim.x)
    tok[c] = t[(size_t)row * DIM + c];
  __syncthreads();
  for (int j = threadIdx.x; j < NPOSE; j += blockDim.x) {
    float s = bp[j] + ip[j];
    for (int k = 0; k < DIM; ++k) s += tok[k] * wp[(size_t)k * NPOSE + j];
    pose[j] = s;
  }
  for (int j = threadIdx.x; j < 10; j += blockDim.x) {
    float s = bsh[j] + ib[j];
    for (int k = 0; k < DIM; ++k) s += tok[k] * wsh[(size_t)k * 10 + j];
    betas[j] = s;
    float s2 = be[j] + ie[j];
    for (int k = 0; k < DIM; ++k) s2 += tok[k] * we[(size_t)k * 10 + j];
    expr[j] = s2;
  }
  if (threadIdx.x < 3) {
    int j = threadIdx.x;
    float s = bc[j] + ic[j];
    for (int k = 0; k < DIM; ++k) s += tok[k] * wc[(size_t)k * 3 + j];
    cam[j] = s;
  }
  __syncthreads();
  float* o = out + (size_t)d * OUTD;
  for (int j = threadIdx.x; j < NROT; j += blockDim.x) {
    float a1x = pose[j * 6 + 0], a1y = pose[j * 6 + 1], a1z = pose[j * 6 + 2];
    float a2x = pose[j * 6 + 3], a2y = pose[j * 6 + 4], a2z = pose[j * 6 + 5];
    float n1 = rsqrtf(a1x * a1x + a1y * a1y + a1z * a1z);
    float b1x = a1x * n1, b1y = a1y * n1, b1z = a1z * n1;
    float dp = b1x * a2x + b1y * a2y + b1z * a2z;
    float c2x = a2x - dp * b1x, c2y = a2y - dp * b1y, c2z = a2z - dp * b1z;
    float n2 = rsqrtf(c2x * c2x + c2y * c2y + c2z * c2z);
    float b2x = c2x * n2, b2y = c2y * n2, b2z = c2z * n2;
    float b3x = b1y * b2z - b1z * b2y;
    float b3y = b1z * b2x - b1x * b2z;
    float b3z = b1x * b2y - b1y * b2x;
    o[j * 9 + 0] = b1x; o[j * 9 + 1] = b2x; o[j * 9 + 2] = b3x;
    o[j * 9 + 3] = b1y; o[j * 9 + 4] = b2y; o[j * 9 + 5] = b3y;
    o[j * 9 + 6] = b1z; o[j * 9 + 7] = b2z; o[j * 9 + 8] = b3z;
  }
  for (int j = threadIdx.x; j < 10; j += blockDim.x) {
    o[477 + j] = betas[j];
    o[487 + j] = expr[j];
  }
  if (threadIdx.x < 3) o[497 + threadIdx.x] = cam[threadIdx.x];
}

// ---------------------------------------------------------------------------
// host
// ---------------------------------------------------------------------------
extern "C" void kernel_launch(void* const* d_in, const int* in_sizes, int n_in,
                              void* d_out, int out_size, void* d_ws, size_t ws_size,
                              hipStream_t stream) {
  (void)in_sizes; (void)n_in; (void)out_size; (void)ws_size;
  // --- input mapping: top dict insertion order; params as JAX pytree leaves
  const float* x_central = (const float*)d_in[0];
  const float* x         = (const float*)d_in[1];
  const int*   idx0      = (const int*)d_in[2];
  const int*   idxy      = (const int*)d_in[3];
  const int*   idxx      = (const int*)d_in[4];
  const float* cq_x      = (const float*)d_in[5];
  const float* cq_y      = (const float*)d_in[6];
  const float* cv_x      = (const float*)d_in[7];
  const float* cv_y      = (const float*)d_in[8];
  const float* deccam_b  = (const float*)d_in[9];
  const float* deccam_w  = (const float*)d_in[10];
  const float* decexpr_b = (const float*)d_in[11];
  const float* decexpr_w = (const float*)d_in[12];
  const float* decpose_b = (const float*)d_in[13];
  const float* decpose_w = (const float*)d_in[14];
  const float* decshape_b= (const float*)d_in[15];
  const float* decshape_w= (const float*)d_in[16];
  const float* init_betas= (const float*)d_in[17];
  const float* init_cam  = (const float*)d_in[18];
  const float* init_expr = (const float*)d_in[19];
  const float* init_pose = (const float*)d_in[20];
  struct Layer {
    const float *attn_out_b, *attn_out_w, *ff1_b, *ff1_w, *ff2_b, *ff2_w,
                *kv_w, *ln1_b, *ln1_g, *ln2_b, *ln2_g, *ln3_b, *ln3_g,
                *q_w, *qkv_w, *xattn_out_b, *xattn_out_w;
  } L[2];
  for (int l = 0; l < 2; ++l) {
    int base = 21 + l * 17;
    L[l].attn_out_b  = (const float*)d_in[base + 0];
    L[l].attn_out_w  = (const float*)d_in[base + 1];
    L[l].ff1_b       = (const float*)d_in[base + 2];
    L[l].ff1_w       = (const float*)d_in[base + 3];
    L[l].ff2_b       = (const float*)d_in[base + 4];
    L[l].ff2_w       = (const float*)d_in[base + 5];
    L[l].kv_w        = (const float*)d_in[base + 6];
    L[l].ln1_b       = (const float*)d_in[base + 7];
    L[l].ln1_g       = (const float*)d_in[base + 8];
    L[l].ln2_b       = (const float*)d_in[base + 9];
    L[l].ln2_g       = (const float*)d_in[base + 10];
    L[l].ln3_b       = (const float*)d_in[base + 11];
    L[l].ln3_g       = (const float*)d_in[base + 12];
    L[l].q_w         = (const float*)d_in[base + 13];
    L[l].qkv_w       = (const float*)d_in[base + 14];
    L[l].xattn_out_b = (const float*)d_in[base + 15];
    L[l].xattn_out_w = (const float*)d_in[base + 16];
  }
  const float* posemb = (const float*)d_in[55];
  const float* tok_b  = (const float*)d_in[56];
  const float* tok_w  = (const float*)d_in[57];

  // --- workspace carving
  char* p = (char*)d_ws;
  auto alloc = [&](size_t bytes) -> char* {
    char* r = p;
    p += (bytes + 255) & ~(size_t)255;
    return r;
  };
  int*    ws_pos    = (int*)alloc(N_DET * 4);
  int*    ws_counts = (int*)alloc(N_IMG * 4);
  int*    ws_ends   = (int*)alloc(N_IMG * 4);
  float*  bias_tok  = (float*)alloc(DIM * 4);
  half_t* ctx_h     = (half_t*)alloc((size_t)N_IMG * HW * CTX * 2);
  half_t* token_h   = (half_t*)alloc((size_t)N_DET * TOKPAD * 2);
  float*  t_f       = (float*)alloc((size_t)N_DET * DIM * 4);
  half_t* ln_h      = (half_t*)alloc((size_t)N_DET * DIM * 2);
  half_t* qkv_h     = (half_t*)alloc((size_t)N_DET * 3 * INNER * 2);
  half_t* attni_h   = (half_t*)alloc((size_t)N_DET * INNER * 2);
  half_t* q_h       = (half_t*)alloc((size_t)N_DET * INNER * 2);
  half_t* kv_h      = (half_t*)alloc((size_t)N_IMG * HW * 2 * INNER * 2);
  half_t* xattni_h  = (half_t*)alloc((size_t)N_DET * INNER * 2);
  half_t* ff1_h     = (half_t*)alloc((size_t)N_DET * MLP * 2);
  half_t* tokw_h    = (half_t*)alloc((size_t)TOKPAD * DIM * 2);
  half_t *w_qkv[2], *w_attn[2], *w_q[2], *w_kv[2], *w_xattn[2], *w_ff1[2], *w_ff2[2];
  for (int l = 0; l < 2; ++l) {
    w_qkv[l]   = (half_t*)alloc((size_t)DIM * 3 * INNER * 2);
    w_attn[l]  = (half_t*)alloc((size_t)INNER * DIM * 2);
    w_q[l]     = (half_t*)alloc((size_t)DIM * INNER * 2);
    w_kv[l]    = (half_t*)alloc((size_t)CTX * 2 * INNER * 2);
    w_xattn[l] = (half_t*)alloc((size_t)INNER * DIM * 2);
    w_ff1[l]   = (half_t*)alloc((size_t)DIM * MLP * 2);
    w_ff2[l]   = (half_t*)alloc((size_t)MLP * DIM * 2);
  }

  auto conv = [&](half_t* dst, const float* src, long nsrc, long ntot) {
    unsigned blocks = (unsigned)((ntot + 255) / 256);
    convert_f16<<<dim3(blocks), dim3(256), 0, stream>>>(dst, src, nsrc, ntot);
  };
  auto gemm = [&](const half_t* A, const half_t* B, const float* bias, const float* res,
                  float* Cf, half_t* Ch, int M, int N, int K, int gelu) {
    dim3 grid(M / 32, (N + 127) / 128);
    wmma_gemm<<<grid, dim3(128), 0, stream>>>(A, B, bias, res, Cf, Ch, M, N, K, gelu);
  };

  // --- stage 0: prep + weight conversion + builders
  prep_kernel<<<dim3(1), dim3(N_DET), 0, stream>>>(idx0, ws_pos, ws_counts, ws_ends);
  conv(tokw_h, tok_w, (long)TOKDIM * DIM, (long)TOKPAD * DIM);
  for (int l = 0; l < 2; ++l) {
    conv(w_qkv[l],   L[l].qkv_w,       (long)DIM * 3 * INNER, (long)DIM * 3 * INNER);
    conv(w_attn[l],  L[l].attn_out_w,  (long)INNER * DIM,     (long)INNER * DIM);
    conv(w_q[l],     L[l].q_w,         (long)DIM * INNER,     (long)DIM * INNER);
    conv(w_kv[l],    L[l].kv_w,        (long)CTX * 2 * INNER, (long)CTX * 2 * INNER);
    conv(w_xattn[l], L[l].xattn_out_w, (long)INNER * DIM,     (long)INNER * DIM);
    conv(w_ff1[l],   L[l].ff1_w,       (long)DIM * MLP,       (long)DIM * MLP);
    conv(w_ff2[l],   L[l].ff2_w,       (long)MLP * DIM,       (long)MLP * DIM);
  }
  combine_bias<<<dim3(4), dim3(256), 0, stream>>>(bias_tok, tok_b, posemb);
  build_token<<<dim3(N_DET), dim3(256), 0, stream>>>(
      x_central, cq_x, cq_y, idx0, idxy, idxx, ws_pos,
      init_pose, init_betas, init_cam, token_h);
  build_ctx<<<dim3(N_IMG, CTX), dim3(HW), 0, stream>>>(
      x, cv_x, cv_y, idxy, idxx, ws_counts, ws_ends, ctx_h);

  // --- token embed: t = token @ tok.w + (tok.b + pos)
  gemm(token_h, tokw_h, bias_tok, nullptr, t_f, nullptr, N_DET, DIM, TOKPAD, 0);

  // --- transformer layers
  for (int l = 0; l < 2; ++l) {
    layernorm_f16<<<dim3(N_DET), dim3(256), 0, stream>>>(t_f, L[l].ln1_g, L[l].ln1_b, ln_h);
    gemm(ln_h, w_qkv[l], nullptr, nullptr, nullptr, qkv_h, N_DET, 3 * INNER, DIM, 0);
    self_attn<<<dim3(N_IMG * HEADS), dim3(64), 0, stream>>>(qkv_h, ws_counts, attni_h);
    gemm(attni_h, w_attn[l], L[l].attn_out_b, t_f, t_f, nullptr, N_DET, DIM, INNER, 0);

    layernorm_f16<<<dim3(N_DET), dim3(256), 0, stream>>>(t_f, L[l].ln2_g, L[l].ln2_b, ln_h);
    gemm(ln_h, w_q[l], nullptr, nullptr, nullptr, q_h, N_DET, INNER, DIM, 0);
    gemm(ctx_h, w_kv[l], nullptr, nullptr, nullptr, kv_h, N_IMG * HW, 2 * INNER, CTX, 0);
    cross_attn<<<dim3(N_IMG * HEADS), dim3(256), 0, stream>>>(q_h, kv_h, xattni_h);
    gemm(xattni_h, w_xattn[l], L[l].xattn_out_b, t_f, t_f, nullptr, N_DET, DIM, INNER, 0);

    layernorm_f16<<<dim3(N_DET), dim3(256), 0, stream>>>(t_f, L[l].ln3_g, L[l].ln3_b, ln_h);
    gemm(ln_h, w_ff1[l], L[l].ff1_b, nullptr, nullptr, ff1_h, N_DET, MLP, DIM, 1);
    gemm(ff1_h, w_ff2[l], L[l].ff2_b, t_f, t_f, nullptr, N_DET, DIM, MLP, 0);
  }

  // --- decode heads + rot6d -> [128, 500]
  decode_kernel<<<dim3(N_DET), dim3(128), 0, stream>>>(
      t_f, idx0, ws_pos,
      decpose_w, decpose_b, init_pose,
      decshape_w, decshape_b, init_betas,
      deccam_w, deccam_b, init_cam,
      decexpr_w, decexpr_b, init_expr,
      (float*)d_out);
}
